// _InterpretableMultiHeadAttention_66700842107098
// MI455X (gfx1250) — matrix operand order repaired and checked by reading, not compiled, not run on hardware
//
#include <hip/hip_runtime.h>
#include <hip/hip_bf16.h>
#include <cstdint>

// Problem constants (from reference): B=4, S=2048, D=1024, H=8, dk=128
#define BB 4
#define SS 2048
#define DD 1024
#define HH 8
#define DK 128

// Tiling: each 256-thread block (8 wave32) computes a 128x128 f32 tile.
// Each wave owns one 16-row M-subtile and iterates the 8 N-subtiles.
// LDS tiles are double-buffered so staging of step k+1 (async DMA / ds stores)
// overlaps the WMMAs of step k.
#define BM 128
#define BK 32
#define LDT 40   // LDS row pitch in halves (80 bytes: 16B-aligned, conflict-skewed)

typedef __attribute__((ext_vector_type(16))) _Float16 v16h;
typedef __attribute__((ext_vector_type(8)))  float    v8f;
typedef __attribute__((ext_vector_type(2)))  float    f32x2;

union frag16 { v16h h; uint32_t u[8]; };

// ---- LDS staging ------------------------------------------------------------
// Stage a 128x32 tile of f32 -> f16 into LDS. Loads batched as f32x2 (b64
// clause, one wait), then packed-f16 dword stores. NT: non-temporal loads
// for stream-once sources (the 512MB attn tensor).
template <bool NT>
__device__ inline void stage_f32_t(const float* __restrict__ src, long rowStride,
                                   _Float16* __restrict__ dst) {
  int tid = threadIdx.x;
  f32x2 tmp[8];
#pragma unroll
  for (int j = 0; j < 8; ++j) {
    int idx = tid + j * 256;               // 2048 float2 chunks
    int r = idx >> 4, c2 = idx & 15;
    const f32x2* p = ((const f32x2*)(src + (long)r * rowStride)) + c2;
    tmp[j] = NT ? __builtin_nontemporal_load(p) : *p;
  }
#pragma unroll
  for (int j = 0; j < 8; ++j) {
    int idx = tid + j * 256;
    int r = idx >> 4, c2 = idx & 15;
    union { uint32_t u; _Float16 h[2]; } pk;
    pk.h[0] = (_Float16)tmp[j].x;
    pk.h[1] = (_Float16)tmp[j].y;
    ((uint32_t*)(dst + r * LDT))[c2] = pk.u;
  }
}

// Stage a 128x32 f16 tile with the CDNA5 async global->LDS path (no VGPR data
// movement; tracked by ASYNCcnt). Each lane issues b128 copies.
__device__ inline void stage_f16_async(const _Float16* __restrict__ src,
                                       long rowStrideHalves,
                                       _Float16* __restrict__ dst) {
  int tid = threadIdx.x;
#pragma unroll
  for (int j = 0; j < 2; ++j) {
    int idx = tid + j * 256;               // 512 16-byte chunks (128 rows x 4)
    int r = idx >> 2, c4 = idx & 3;
    uint64_t gaddr = (uint64_t)(uintptr_t)(src + (long)r * rowStrideHalves + c4 * 8);
    uint32_t laddr = (uint32_t)(uintptr_t)(dst + r * LDT + c4 * 8);
    asm volatile("global_load_async_to_lds_b128 %0, %1, off"
                 :: "v"(laddr), "v"(gaddr)
                 : "memory");
  }
}

__device__ inline void wait_async() {
  asm volatile("s_wait_asynccnt 0x0" ::: "memory");
}

// ---- WMMA fragment loads (per ISA 7.12.2 layouts, wave32) -------------------
// A 16x32 f16: lane l holds row (l&15); halves i -> K = (i%8) + 16*(i/8) + 8*(l/16)
__device__ inline v16h load_frag_a(const _Float16* lds, int row, int lane) {
  const uint32_t* p = (const uint32_t*)lds;
  frag16 f;
  int khi = (lane >> 4) << 3;                 // 0 or 8
  int basew = (row * LDT + khi) >> 1;         // dword index (LDT even, khi even)
#pragma unroll
  for (int q = 0; q < 4; ++q) {
    f.u[q]     = p[basew + q];                // K = khi + 0..7
    f.u[q + 4] = p[basew + 8 + q];            // K = khi + 16..23
  }
  return f.h;
}

// B 32x16 f16, n-major LDS tile (row = n): lane l holds col (l&15);
// halves i -> K = (i%8) + 16*(i/8) + 8*(l/16)
__device__ inline v16h load_frag_b(const _Float16* lds, int row, int lane) {
  const uint32_t* p = (const uint32_t*)lds;
  frag16 f;
  int khi = (lane >> 4) << 3;                 // 0 or 8
  int basew = (row * LDT + khi) >> 1;
#pragma unroll
  for (int q = 0; q < 4; ++q) { f.u[q] = p[basew + q]; f.u[q + 4] = p[basew + 8 + q]; }
  return f.h;
}

// One K-step: prefetch A frag + all 8 B frags (single ds clause, one wait),
// then 8 back-to-back WMMAs on independent accumulators.
__device__ inline void wmma_tiles(const _Float16* lA, const _Float16* lB,
                                  int wave, int lane, v8f acc[8]) {
  v16h a = load_frag_a(lA, wave * 16 + (lane & 15), lane);
  v16h b[8];
#pragma unroll
  for (int nt = 0; nt < 8; ++nt) b[nt] = load_frag_b(lB, nt * 16 + (lane & 15), lane);
#pragma unroll
  for (int nt = 0; nt < 8; ++nt) {
    acc[nt] = __builtin_amdgcn_wmma_f32_16x16x32_f16(
        false, a, false, b[nt], (short)0, acc[nt], false, false);
  }
}

// ---- Kernel 1: Q/K/V projections -------------------------------------------
// grid.x = (B*S)/128 = 64 ; grid.y = 17 (0..7 Q heads, 8..15 K heads, 16 = V)
__global__ __launch_bounds__(256) void proj_kernel(
    const float* __restrict__ q, const float* __restrict__ v, const float* __restrict__ k,
    const float* __restrict__ Wq, const float* __restrict__ bq,
    const float* __restrict__ Wk, const float* __restrict__ bk,
    const float* __restrict__ Wv, const float* __restrict__ bv,
    _Float16* __restrict__ qs, _Float16* __restrict__ ksp, _Float16* __restrict__ vsT) {
  __shared__ alignas(16) _Float16 lA[2][BM * LDT];
  __shared__ alignas(16) _Float16 lB[2][BM * LDT];
  int pidx = blockIdx.y;
  const float *X, *W, *bias;
  if (pidx < HH)          { X = q; W = Wq + (long)pidx * DK * DD;        bias = bq + pidx * DK; }
  else if (pidx < 2 * HH) { X = k; W = Wk + (long)(pidx - HH) * DK * DD; bias = bk + (pidx - HH) * DK; }
  else                    { X = v; W = Wv;                               bias = bv; }

  int blockM = blockIdx.x * BM;
  int lane = threadIdx.x & 31, wave = threadIdx.x >> 5;
  v8f acc[8] = {};

  stage_f32_t<false>(X + (long)blockM * DD, DD, lA[0]);
  stage_f32_t<false>(W, DD, lB[0]);
  __syncthreads();
  int buf = 0;
  for (int k0 = 0; k0 < DD; k0 += BK, buf ^= 1) {
    int nk = k0 + BK;
    if (nk < DD) {
      stage_f32_t<false>(X + (long)blockM * DD + nk, DD, lA[buf ^ 1]);
      stage_f32_t<false>(W + nk, DD, lB[buf ^ 1]);
    }
    wmma_tiles(lA[buf], lB[buf], wave, lane, acc);
    __syncthreads();
  }

  int rbase = wave * 16 + ((lane >> 4) << 3);
  int col = lane & 15;
#pragma unroll
  for (int nt = 0; nt < 8; ++nt) {
    int n = nt * 16 + col;
    float bn = bias[n];
#pragma unroll
    for (int r = 0; r < 8; ++r) {
      int m = blockM + rbase + r;
      int b = m >> 11, s = m & (SS - 1);
      float val = acc[nt][r] + bn;
      if (pidx < 2 * HH) {
        _Float16* dst = (pidx < HH) ? qs : ksp;
        int h = (pidx < HH) ? pidx : (pidx - HH);
        dst[(((long)(b * HH + h) * SS + s) << 7) + n] = (_Float16)val;
      } else {
        vsT[((long)b * DK + n) * SS + s] = (_Float16)val;  // transposed: [b][e][t]
      }
    }
  }
}

// ---- Kernel 2: scores = qs . ks^T * 1/sqrt(dk) -> attn region (raw) --------
// grid = (S/128, S/128, B*H)
__global__ __launch_bounds__(256) void scores_kernel(
    const _Float16* __restrict__ qs, const _Float16* __restrict__ ksp,
    float* __restrict__ attn) {
  __shared__ alignas(16) _Float16 lA[2][BM * LDT];
  __shared__ alignas(16) _Float16 lB[2][BM * LDT];
  int bh = blockIdx.z;
  int b = bh >> 3, h = bh & 7;
  const _Float16* A  = qs  + (long)bh * SS * DK;
  const _Float16* Bm = ksp + (long)bh * SS * DK;
  int blockM = blockIdx.x * BM;   // query rows s
  int blockN = blockIdx.y * BM;   // key rows t
  int lane = threadIdx.x & 31, wave = threadIdx.x >> 5;
  v8f acc[8] = {};

  stage_f16_async(A  + (long)blockM * DK, DK, lA[0]);
  stage_f16_async(Bm + (long)blockN * DK, DK, lB[0]);
  wait_async();
  __syncthreads();
  int buf = 0;
  for (int k0 = 0; k0 < DK; k0 += BK, buf ^= 1) {
    int nk = k0 + BK;
    if (nk < DK) {
      stage_f16_async(A  + (long)blockM * DK + nk, DK, lA[buf ^ 1]);
      stage_f16_async(Bm + (long)blockN * DK + nk, DK, lB[buf ^ 1]);
    }
    wmma_tiles(lA[buf], lB[buf], wave, lane, acc);
    wait_async();
    __syncthreads();
  }

  const float scale = 0.08838834764831845f;  // 1/sqrt(128)
  int rbase = wave * 16 + ((lane >> 4) << 3);
  int col = lane & 15;
#pragma unroll
  for (int nt = 0; nt < 8; ++nt) {
#pragma unroll
    for (int r = 0; r < 8; ++r) {
      int s = blockM + rbase + r;
      int t = blockN + nt * 16 + col;
      // stream-once 512MB tensor: non-temporal store, layout [B,S,H,S]
      __builtin_nontemporal_store(acc[nt][r] * scale,
                                  attn + (((long)(b * SS + s)) * HH + h) * SS + t);
    }
  }
}

// ---- Kernel 3: row softmax in place over last axis (len S) -----------------
__global__ __launch_bounds__(256) void softmax_kernel(float* __restrict__ attn) {
  __shared__ float red[256];
  long base = (long)blockIdx.x * SS;
  int tid = threadIdx.x;
  float vals[8];
  float m = -3.0e38f;
#pragma unroll
  for (int j = 0; j < 8; ++j) {
    vals[j] = __builtin_nontemporal_load(attn + base + tid + j * 256);
    m = fmaxf(m, vals[j]);
  }
  red[tid] = m; __syncthreads();
  for (int off = 128; off > 0; off >>= 1) {
    if (tid < off) red[tid] = fmaxf(red[tid], red[tid + off]);
    __syncthreads();
  }
  float rmax = red[0]; __syncthreads();
  float sum = 0.f;
#pragma unroll
  for (int j = 0; j < 8; ++j) { vals[j] = __expf(vals[j] - rmax); sum += vals[j]; }
  red[tid] = sum; __syncthreads();
  for (int off = 128; off > 0; off >>= 1) {
    if (tid < off) red[tid] += red[tid + off];
    __syncthreads();
  }
  float inv = 1.0f / red[0];
#pragma unroll
  for (int j = 0; j < 8; ++j)
    __builtin_nontemporal_store(vals[j] * inv, attn + base + tid + j * 256);
}

// ---- Kernel 4: hm[b,s,e] = (1/H) * sum_h sum_t attn[b,s,h,t] * vsT[b,e,t] --
// grid = (S/128, 1, B); N = dk = 128 exactly one block-wide tile.
// h and t are both reduction dims: flattened into one pipelined K loop.
__global__ __launch_bounds__(256) void av_kernel(
    const float* __restrict__ attn, const _Float16* __restrict__ vsT,
    _Float16* __restrict__ hm) {
  __shared__ alignas(16) _Float16 lA[2][BM * LDT];
  __shared__ alignas(16) _Float16 lB[2][BM * LDT];
  int b = blockIdx.z;
  int blockM = blockIdx.x * BM;  // s rows
  int lane = threadIdx.x & 31, wave = threadIdx.x >> 5;
  v8f acc[8] = {};

  const int NSTEP = HH * (SS / BK);  // 512
  auto stage_step = [&](int idx, _Float16* dA, _Float16* dB) {
    int h = idx >> 6;                 // SS/BK = 64 steps per head
    int t0 = (idx & 63) << 5;
    stage_f16_async(vsT + (long)b * DK * SS + t0, SS, dB);   // vsT rows [e][t] = n-major B
    stage_f32_t<true>(attn + (((long)(b * SS + blockM)) * HH + h) * SS + t0,
                      (long)HH * SS, dA);                    // NT: attn read once
  };

  stage_step(0, lA[0], lB[0]);
  wait_async();
  __syncthreads();
  int buf = 0;
  for (int idx = 0; idx < NSTEP; ++idx, buf ^= 1) {
    if (idx + 1 < NSTEP) stage_step(idx + 1, lA[buf ^ 1], lB[buf ^ 1]);
    wmma_tiles(lA[buf], lB[buf], wave, lane, acc);
    wait_async();
    __syncthreads();
  }

  const float invH = 1.0f / HH;
  int rbase = wave * 16 + ((lane >> 4) << 3);
  int col = lane & 15;
#pragma unroll
  for (int nt = 0; nt < 8; ++nt) {
#pragma unroll
    for (int r = 0; r < 8; ++r) {
      int s = blockM + rbase + r;
      hm[(((long)(b * SS + s)) << 7) + nt * 16 + col] = (_Float16)(acc[nt][r] * invH);
    }
  }
}

// ---- Kernel 5: out[m, d] = hm[m, :] . Wh[d, :]  (m = b*S+s) ----------------
// grid = ((B*S)/128, D/128)
__global__ __launch_bounds__(256) void out_kernel(
    const _Float16* __restrict__ hm, const float* __restrict__ Wh,
    float* __restrict__ out) {
  __shared__ alignas(16) _Float16 lA[2][BM * LDT];
  __shared__ alignas(16) _Float16 lB[2][BM * LDT];
  int blockM = blockIdx.x * BM;
  int blockN = blockIdx.y * BM;
  int lane = threadIdx.x & 31, wave = threadIdx.x >> 5;
  v8f acc[8] = {};

  stage_f16_async(hm + (long)blockM * DK, DK, lA[0]);
  stage_f32_t<false>(Wh + (long)blockN * DK, DK, lB[0]);
  wait_async();
  __syncthreads();
  int buf = 0;
  for (int k0 = 0; k0 < DK; k0 += BK, buf ^= 1) {
    int nk = k0 + BK;
    if (nk < DK) {
      stage_f16_async(hm + (long)blockM * DK + nk, DK, lA[buf ^ 1]);
      stage_f32_t<false>(Wh + (long)blockN * DK + nk, DK, lB[buf ^ 1]);
    }
    wmma_tiles(lA[buf], lB[buf], wave, lane, acc);
    wait_async();
    __syncthreads();
  }

  int rbase = wave * 16 + ((lane >> 4) << 3);
  int col = lane & 15;
#pragma unroll
  for (int nt = 0; nt < 8; ++nt) {
#pragma unroll
    for (int r = 0; r < 8; ++r) {
      int m = blockM + rbase + r;
      __builtin_nontemporal_store(acc[nt][r],
                                  out + (long)m * DD + blockN + nt * 16 + col);
    }
  }
}

extern "C" void kernel_launch(void* const* d_in, const int* in_sizes, int n_in,
                              void* d_out, int out_size, void* d_ws, size_t ws_size,
                              hipStream_t stream) {
  const float* q  = (const float*)d_in[0];
  const float* v  = (const float*)d_in[1];
  const float* k  = (const float*)d_in[2];
  const float* Wq = (const float*)d_in[3];
  const float* bq = (const float*)d_in[4];
  const float* Wk = (const float*)d_in[5];
  const float* bk = (const float*)d_in[6];
  const float* Wv = (const float*)d_in[7];
  const float* bv = (const float*)d_in[8];
  const float* Wh = (const float*)d_in[9];

  float* out  = (float*)d_out;                    // [B,S,D]
  float* attn = out + (long)BB * SS * DD;         // [B,S,H,S]

  // Workspace layout (f16): qs, ks [B,H,S,dk]; vsT [B,dk,S]; hm [B,S,dk] ~38MB
  _Float16* qs  = (_Float16*)d_ws;
  _Float16* ksp = qs  + (long)BB * HH * SS * DK;
  _Float16* vsT = ksp + (long)BB * HH * SS * DK;
  _Float16* hm  = vsT + (long)BB * DK * SS;

  proj_kernel<<<dim3((BB * SS) / BM, 2 * HH + 1), 256, 0, stream>>>(
      q, v, k, Wq, bq, Wk, bk, Wv, bv, qs, ksp, vsT);
  scores_kernel<<<dim3(SS / BM, SS / BM, BB * HH), 256, 0, stream>>>(qs, ksp, attn);
  softmax_kernel<<<dim3(BB * SS * HH), 256, 0, stream>>>(attn);
  av_kernel<<<dim3(SS / BM, 1, BB), 256, 0, stream>>>(attn, vsT, hm);
  out_kernel<<<dim3((BB * SS) / BM, DD / BM), 256, 0, stream>>>(hm, Wh, out);
}
